// DecoderRNN_25769804062
// MI455X (gfx1250) — matrix operand order repaired and checked by reading, not compiled, not run on hardware
//
#include <hip/hip_runtime.h>
#include <hip/hip_bf16.h>
#include <math.h>

// ---------------------------------------------------------------------------
// DecoderRNN on MI455X (gfx1250, wave32, WMMA):
//   A: weights fp32->bf16, X0 = [embed(tok), features] bf16
//   B: ig = X @ W_ihT + bias            (bf16 WMMA, fp32 accumulate)
//   C: recurrent scan, 32 persistent WGs; W_hh slice (256KB) + H stage (64KB)
//      resident in LDS in pre-swizzled WMMA-fragment layout (320KB WGP LDS);
//      H staged per step via GLOBAL_LOAD_ASYNC_TO_LDS_B128 (ASYNCcnt path);
//      one device-scope barrier per step (ping-pong H buffer).
//   D: logits = H2 @ fc_wT + fc_b -> (B,T,V)   (compute-bound, 134 GFLOP)
// ---------------------------------------------------------------------------

typedef __attribute__((ext_vector_type(16))) __bf16 v16bf;
typedef __attribute__((ext_vector_type(8)))  float  v8f;

// Types for the async-to-LDS builtin: (v4i as1*, v4i as3*, imm, imm)
typedef int v4i_t __attribute__((vector_size(16)));
typedef __attribute__((address_space(1))) v4i_t v4i_as1;
typedef __attribute__((address_space(3))) v4i_t v4i_as3;

#define TT 32
#define BB 64
#define VV 32000
#define KD 1024
#define MR 2048
#define G4 4096

#if defined(__HIP_DEVICE_COMPILE__) && __has_builtin(__builtin_amdgcn_global_load_async_to_lds_b128)
#define HAS_ASYNC_LDS 1
#else
#define HAS_ASYNC_LDS 0
#endif

__device__ __forceinline__ void wait_async0() {
#if defined(__HIP_DEVICE_COMPILE__)
#if __has_builtin(__builtin_amdgcn_s_wait_asynccnt)
  __builtin_amdgcn_s_wait_asynccnt(0);
#else
  asm volatile("s_wait_asynccnt 0" ::: "memory");
#endif
#endif
}

__device__ __forceinline__ unsigned short f2bf(float f) {
  unsigned u = __float_as_uint(f);
  u += 0x7fffu + ((u >> 16) & 1u);  // round-to-nearest-even
  return (unsigned short)(u >> 16);
}

// A fragment (16-bit A 16x32, ISA 7.12.2) loaded straight from row-major bf16.
__device__ __forceinline__ v16bf load_a_frag(const unsigned short* A, int lda,
                                             int m0, int k0, int lane) {
  const unsigned short* p = A + (size_t)(m0 + (lane & 15)) * lda + k0 + ((lane >> 4) << 4);
  v16bf a;
  ((uint4*)&a)[0] = *(const uint4*)p;         // K = base .. base+7
  ((uint4*)&a)[1] = *(const uint4*)(p + 8);   // K = base+8 .. base+15
  return a;
}

// B fragment (32x16, B = W^T tile) from row-major bf16 W.
__device__ __forceinline__ v16bf load_b_frag(const unsigned short* W, int ldw,
                                             int n0, int k0, int lane) {
  const unsigned short* p = W + (size_t)(n0 + (lane & 15)) * ldw + k0 + ((lane >> 4) << 4);
  v16bf b;
  ((uint4*)&b)[0] = *(const uint4*)p;
  ((uint4*)&b)[1] = *(const uint4*)(p + 8);
  return b;
}

// Pre-swizzled fragment in LDS: frag fi = 32 lanes x 16 bf16, lane-contiguous
// 32B blocks -> full-rate ds_load_b128, no padding needed.
__device__ __forceinline__ v16bf load_frag_lds(const unsigned short* base, int fi, int lane) {
  const unsigned short* p = base + (size_t)fi * 512 + lane * 16;
  v16bf a;
  ((uint4*)&a)[0] = *(const uint4*)p;
  ((uint4*)&a)[1] = *(const uint4*)(p + 8);
  return a;
}

__device__ __forceinline__ v8f wmma_bf16(v16bf a, v16bf b, v8f c) {
  return __builtin_amdgcn_wmma_f32_16x16x32_bf16(false, a, false, b, (short)0, c,
                                                 false, false);
}

// Device-scope two-phase barrier: bar[0]=count, bar[1]=generation.
__device__ __forceinline__ void gridbar(unsigned* bar, unsigned nwg) {
  __syncthreads();
  if (threadIdx.x == 0) {
    __threadfence();
    volatile unsigned* vcnt = bar;
    volatile unsigned* vgen = bar + 1;
    unsigned g = *vgen;
    if (atomicAdd((unsigned*)bar, 1u) == nwg - 1) {
      *vcnt = 0;
      __threadfence();
      atomicAdd((unsigned*)(bar + 1), 1u);
    } else {
      while (*vgen == g) { __builtin_amdgcn_s_sleep(8); }
    }
    __threadfence();
  }
  __syncthreads();
}

// ---------------------------------------------------------------------------

__global__ void k_init(unsigned* bar) {
  if (threadIdx.x < 64) bar[threadIdx.x] = 0u;
}

__global__ void k_f32_to_bf16(const float* __restrict__ src,
                              unsigned short* __restrict__ dst, int n) {
  for (int i = blockIdx.x * blockDim.x + threadIdx.x; i < n;
       i += gridDim.x * blockDim.x)
    dst[i] = f2bf(src[i]);
}

__global__ void k_build_x0(const float* __restrict__ features,
                           const int* __restrict__ captions,
                           const float* __restrict__ embed_w,
                           unsigned short* __restrict__ X0) {
  int idx = blockIdx.x * blockDim.x + threadIdx.x;
  if (idx >= MR * KD) return;
  int row = idx >> 10, col = idx & 1023;
  int t = row >> 6, b = row & 63;
  float v;
  if (col < 512) {
    int tok = (t == 0) ? 0 : captions[b * TT + (t - 1)];
    v = embed_w[(size_t)tok * 512 + col];
  } else {
    v = features[b * 512 + (col - 512)];
  }
  X0[idx] = f2bf(v);
}

// OUT[M x Nn] = A(bf16, Mx1024) @ Wb^T(bf16, Nnx1024) + bias1 (+bias2).
// Wave tile 32x64 (acc[2][4]): 12 b128 loads per 8 WMMAs. WG tile 64x256.
__global__ __launch_bounds__(256) void k_gemm(
    const unsigned short* __restrict__ A,
    const unsigned short* __restrict__ Wb,
    const float* __restrict__ bias1,
    const float* __restrict__ bias2,
    float* __restrict__ OUT, int Nn, int logits) {
  int lane = threadIdx.x & 31;
  int w = threadIdx.x >> 5;
  int mw = w >> 2, nw = w & 3;                 // 2x4 wave grid
  int mbase = blockIdx.y * 64 + mw * 32;
  int nbase = blockIdx.x * 256 + nw * 64;
  v8f acc[2][4] = {};
#pragma unroll 2
  for (int kc = 0; kc < 32; ++kc) {
    int k0 = kc * 32;
    v16bf a0 = load_a_frag(A, KD, mbase, k0, lane);
    v16bf a1 = load_a_frag(A, KD, mbase + 16, k0, lane);
    if (kc + 1 < 32)
      __builtin_prefetch(Wb + (size_t)(nbase + (lane & 15)) * KD + k0 + 32, 0, 1);
#pragma unroll
    for (int ni = 0; ni < 4; ++ni) {
      v16bf bf = load_b_frag(Wb, KD, nbase + ni * 16, k0, lane);
      acc[0][ni] = wmma_bf16(a0, bf, acc[0][ni]);
      acc[1][ni] = wmma_bf16(a1, bf, acc[1][ni]);
    }
  }
#pragma unroll
  for (int mi = 0; mi < 2; ++mi)
#pragma unroll
    for (int ni = 0; ni < 4; ++ni)
#pragma unroll
      for (int r = 0; r < 8; ++r) {
        int m = mbase + mi * 16 + r + ((lane >> 4) << 3);
        int n = nbase + ni * 16 + (lane & 15);
        float v = acc[mi][ni][r] + bias1[n] + (bias2 ? bias2[n] : 0.f);
        if (logits) {
          int t = m >> 6, b = m & 63;
          OUT[(size_t)b * (TT * VV) + (size_t)t * VV + n] = v;
        } else {
          OUT[(size_t)m * Nn + n] = v;
        }
      }
}

// Recurrent scan. 32 persistent WGs x 256 thr. WG owns hidden idx nn0..nn0+32
// (4 gates -> 128 W_hh rows). LDS: 256KB weight fragments + 64KB H-stage
// fragments (Glds 16KB aliased over H-stage behind a sync) = 320KB exactly.
#define WFRAGS (8 * 32)                 // n-chunk x kc
#define HFRAGS (32 * 2)                 // kc x m-tile
#define WLDS_BYTES (WFRAGS * 1024)      // 262144
#define HLDS_BYTES (HFRAGS * 1024)      // 65536
#define SCAN_SMEM (WLDS_BYTES + HLDS_BYTES)  // 327680 = 320KB

__global__ __launch_bounds__(256) void k_scan(
    const float* __restrict__ ig,        // 2048 x 4096 fp32 pre-activations
    const float* __restrict__ w_hh,      // 4096 x 1024 fp32
    unsigned short* __restrict__ h_out,  // 2048 x 1024 bf16
    unsigned short* __restrict__ Hmat,   // 2 x 32 x 1024 bf16 ping-pong
    unsigned* __restrict__ bar) {
  extern __shared__ unsigned char smem[];
  unsigned short* WldsU = (unsigned short*)smem;
  unsigned short* HldsU = (unsigned short*)(smem + WLDS_BYTES);
  float* Glds = (float*)(smem + WLDS_BYTES);   // aliases H-stage (sync-guarded)
  int tid = threadIdx.x;
  int lane = tid & 31;
  int w = tid >> 5;
  int nn0 = blockIdx.x * 32;

  // W_hh slice -> LDS as pre-swizzled B fragments (fp32 -> bf16 on the fly).
  // local col lc = g*32+d  <->  W row g*1024+nn0+d ; frag fi = (lc>>4)*32 + kc
  for (int idx = tid; idx < 128 * 1024; idx += 256) {
    int lr = idx >> 10, kk = idx & 1023;
    int g = lr >> 5, d = lr & 31;
    float wv = w_hh[(size_t)(g * 1024 + nn0 + d) * 1024 + kk];
    int nc = lr >> 4, lpart = lr & 15;
    int kc = kk >> 5, rem = kk & 31;
    int lane2 = ((rem >> 4) << 4) | lpart;
    int e = rem & 15;
    WldsU[((size_t)(nc * 32 + kc) * 32 + lane2) * 16 + e] = f2bf(wv);
  }
  for (int q = tid; q < 32 * 32; q += 256) {   // zero our columns of Hmat[0]
    int t = q >> 5, d = q & 31;
    Hmat[t * 1024 + nn0 + d] = 0;
  }
  float c[4] = {0.f, 0.f, 0.f, 0.f};
  gridbar(bar, 32);

  int p = 0;
  for (int step = 0; step < 64; ++step) {
    const unsigned short* Hp = Hmat + p * (32 * 1024);
    // Stage Hmat[p] -> LDS A-fragments via async-to-LDS (ASYNCcnt path).
    // frag fi = kc*2 + mt; per lane two 16B pieces (K halves of the A layout).
#pragma unroll
    for (int f = 0; f < 8; ++f) {
      int fi = w * 8 + f;
      int kc = fi >> 1, mt = fi & 1;
      const unsigned short* gp =
          Hp + (size_t)(mt * 16 + (lane & 15)) * 1024 + kc * 32 + ((lane >> 4) << 4);
      unsigned short* lp = HldsU + (size_t)fi * 512 + lane * 16;
#if HAS_ASYNC_LDS
      __builtin_amdgcn_global_load_async_to_lds_b128(
          (v4i_as1*)gp, (v4i_as3*)lp, 0, 0);
      __builtin_amdgcn_global_load_async_to_lds_b128(
          (v4i_as1*)(gp + 8), (v4i_as3*)(lp + 8), 0, 0);
#else
      *(uint4*)lp = *(const uint4*)gp;
      *(uint4*)(lp + 8) = *(const uint4*)(gp + 8);
#endif
    }
#if HAS_ASYNC_LDS
    wait_async0();
#endif
    __syncthreads();

    int n0l = w * 16;                    // wave owns 16 local gate columns
    v8f acc[2] = {v8f{}, v8f{}};
#pragma unroll 4
    for (int kc = 0; kc < 32; ++kc) {
      v16bf bf = load_frag_lds(WldsU, w * 32 + kc, lane);
      v16bf a0 = load_frag_lds(HldsU, kc * 2 + 0, lane);
      v16bf a1 = load_frag_lds(HldsU, kc * 2 + 1, lane);
      acc[0] = wmma_bf16(a0, bf, acc[0]);
      acc[1] = wmma_bf16(a1, bf, acc[1]);
    }
    __syncthreads();   // all H-stage reads done before Glds aliases the region

    // pre-activation = recurrent term + input term
#pragma unroll
    for (int mt = 0; mt < 2; ++mt)
#pragma unroll
      for (int r = 0; r < 8; ++r) {
        int t = mt * 16 + r + ((lane >> 4) << 3);
        int lc = n0l + (lane & 15);                // gate = lc>>5, d = lc&31
        int j = ((lc >> 5) << 10) + nn0 + (lc & 31);
        Glds[t * 128 + lc] = acc[mt][r] + ig[(size_t)(t * 64 + step) * G4 + j];
      }
    __syncthreads();

    // fused gate math; cell state in registers (fixed (t,d) per thread)
    unsigned short* Hn = Hmat + (p ^ 1) * (32 * 1024);
#pragma unroll
    for (int pq = 0; pq < 4; ++pq) {
      int q = tid * 4 + pq;
      int t = q >> 5, d = q & 31;
      float gi = Glds[t * 128 + d];
      float gf = Glds[t * 128 + 32 + d];
      float gg = Glds[t * 128 + 64 + d];
      float go = Glds[t * 128 + 96 + d];
      gi = 1.f / (1.f + __expf(-gi));
      gf = 1.f / (1.f + __expf(-gf));
      gg = tanhf(gg);
      go = 1.f / (1.f + __expf(-go));
      float cc = gf * c[pq] + gi * gg;
      c[pq] = cc;
      float hh = go * tanhf(cc);
      unsigned short hb = f2bf(hh);
      Hn[t * 1024 + nn0 + d] = hb;
      h_out[(size_t)(t * 64 + step) * 1024 + nn0 + d] = hb;
    }
    gridbar(bar, 32);  // publish h (double buffer -> one barrier per step)
    p ^= 1;
  }
}

// ---------------------------------------------------------------------------

extern "C" void kernel_launch(void* const* d_in, const int* in_sizes, int n_in,
                              void* d_out, int out_size, void* d_ws, size_t ws_size,
                              hipStream_t stream) {
  (void)in_sizes; (void)n_in; (void)out_size; (void)ws_size;
  const float* features = (const float*)d_in[0];
  const int*   captions = (const int*)d_in[1];
  const float* embed_w  = (const float*)d_in[2];
  const float* w_ih0    = (const float*)d_in[3];
  const float* w_hh0    = (const float*)d_in[4];
  const float* b_ih0    = (const float*)d_in[5];
  const float* b_hh0    = (const float*)d_in[6];
  const float* w_ih1    = (const float*)d_in[7];
  const float* w_hh1    = (const float*)d_in[8];
  const float* b_ih1    = (const float*)d_in[9];
  const float* b_hh1    = (const float*)d_in[10];
  const float* fc_w     = (const float*)d_in[11];
  const float* fc_b     = (const float*)d_in[12];
  float* out = (float*)d_out;

  char* ws = (char*)d_ws;
  size_t off = 0;
  auto take = [&](size_t bytes) {
    char* p = ws + off;
    off += (bytes + 255) & ~(size_t)255;
    return p;
  };
  unsigned short* wih0b = (unsigned short*)take((size_t)G4 * KD * 2);
  unsigned short* wih1b = (unsigned short*)take((size_t)G4 * KD * 2);
  unsigned short* fcwb  = (unsigned short*)take((size_t)VV * KD * 2);
  unsigned short* X0    = (unsigned short*)take((size_t)MR * KD * 2);
  unsigned short* H1    = (unsigned short*)take((size_t)MR * KD * 2);
  unsigned short* H2    = (unsigned short*)take((size_t)MR * KD * 2);
  float*          IG    = (float*)take((size_t)MR * G4 * 4);
  unsigned short* HM    = (unsigned short*)take((size_t)2 * 32 * 1024 * 2);
  unsigned*       BAR   = (unsigned*)take(256);

  (void)hipFuncSetAttribute((const void*)k_scan,
                            hipFuncAttributeMaxDynamicSharedMemorySize, SCAN_SMEM);

  k_init<<<1, 64, 0, stream>>>(BAR);
  k_f32_to_bf16<<<2048, 256, 0, stream>>>(w_ih0, wih0b, G4 * KD);
  k_f32_to_bf16<<<2048, 256, 0, stream>>>(w_ih1, wih1b, G4 * KD);
  k_f32_to_bf16<<<4096, 256, 0, stream>>>(fc_w, fcwb, VV * KD);
  k_build_x0<<<(MR * KD) / 256, 256, 0, stream>>>(features, captions, embed_w, X0);

  // layer 0: input GEMM + scan
  k_gemm<<<dim3(G4 / 256, MR / 64), 256, 0, stream>>>(X0, wih0b, b_ih0, b_hh0, IG, G4, 0);
  k_scan<<<32, 256, SCAN_SMEM, stream>>>(IG, w_hh0, H1, HM, BAR);

  // layer 1: input GEMM (reuse IG) + scan
  k_gemm<<<dim3(G4 / 256, MR / 64), 256, 0, stream>>>(H1, wih1b, b_ih1, b_hh1, IG, G4, 0);
  k_scan<<<32, 256, SCAN_SMEM, stream>>>(IG, w_hh1, H2, HM, BAR + 8);

  // logits: (B,T,V) scatter
  k_gemm<<<dim3(VV / 256, MR / 64), 256, 0, stream>>>(H2, fcwb, fc_b, nullptr, out, VV, 1);
}
